// MultiHeadAttention_27453430956540
// MI455X (gfx1250) — matrix operand order repaired and checked by reading, not compiled, not run on hardware
//
#include <hip/hip_runtime.h>

typedef _Float16 half_t;
typedef __attribute__((ext_vector_type(16))) _Float16 v16h;
typedef __attribute__((ext_vector_type(8)))  float    v8f;
typedef __attribute__((ext_vector_type(8)))  unsigned int v8u;

static constexpr int Bb  = 8;
static constexpr int Nn  = 1024;
static constexpr int DIM = 768;
static constexpr int H   = 12;
static constexpr int Dh  = 64;
static constexpr float SCALE = 0.125f;              // 1/sqrt(64)
static constexpr float LOG2E = 1.4426950408889634f; // fold into Q -> exp2 softmax

#define WMMA(a, b, c) __builtin_amdgcn_wmma_f32_16x16x32_f16( \
    false, (a), false, (b), (short)0, (c), false, false)

// ---- fragment loaders (contiguous k order: lane half h holds k 16h..16h+15, wave32) -----------------

// A-matrix 16x32 f16. This lane's row = rowBase + (lane&15); lanes>=16 carry
// K 16..31. Consecutive K pairs are contiguous -> dword loads.
__device__ __forceinline__ v16h loadA(const half_t* A, int ld, int row, int k0, int lane) {
    const int hi = (lane >> 4) * 16;
    const half_t* r = A + (size_t)row * ld + k0;
    v8u u;
#pragma unroll
    for (int v = 0; v < 8; ++v) {
        int kb = 2 * v + hi;
        u[v] = *(const unsigned int*)(r + kb);
    }
    return __builtin_bit_cast(v16h, u);
}

// B-matrix 32x16 f16 where B[k][n] = S[n][k] (S row-major, contraction dim
// contiguous). VGPR v packs rows khalf+2v, khalf+2v+1; column = ncol.
__device__ __forceinline__ v16h loadBT(const half_t* S, int ld, int ncol, int k0, int lane) {
    const half_t* r = S + (size_t)ncol * ld + k0 + ((lane >> 4) * 16);
    v8u u;
#pragma unroll
    for (int v = 0; v < 8; ++v) u[v] = *(const unsigned int*)(r + 2 * v);
    return __builtin_bit_cast(v16h, u);
}

// B-matrix 32x16 f16 from a genuinely row-major B[k][n] (strided in k).
__device__ __forceinline__ v16h loadBS(const half_t* Bm, int ld, int k0, int ncol, int lane) {
    const half_t* p = Bm + (size_t)(k0 + ((lane >> 4) * 16)) * ld + ncol;
    union { half_t h[2]; unsigned int w; } pk;
    v8u u;
#pragma unroll
    for (int v = 0; v < 8; ++v) {
        pk.h[0] = p[(2 * v) * ld];
        pk.h[1] = p[(2 * v + 1) * ld];
        u[v] = pk.w;
    }
    return __builtin_bit_cast(v16h, u);
}

// ---- kernel 0: fp32 -> f16 conversion ---------------------------------------

__global__ void mha_cvt(const float* __restrict__ x, const float* __restrict__ wq,
                        const float* __restrict__ wp, half_t* __restrict__ xh,
                        half_t* __restrict__ wqh, half_t* __restrict__ wph) {
    const int NX = Bb * Nn * DIM, NQ = 3 * DIM * DIM, NP = DIM * DIM;
    const int total = NX + NQ + NP;
    for (int i = blockIdx.x * blockDim.x + threadIdx.x; i < total;
         i += gridDim.x * blockDim.x) {
        if (i < NX)            xh[i]            = (half_t)x[i];
        else if (i < NX + NQ)  wqh[i - NX]      = (half_t)wq[i - NX];
        else                   wph[i - NX - NQ] = (half_t)wp[i - NX - NQ];
    }
}

// ---- kernel 1: QKV projection, scatter to head-major Q/K/V ------------------
// C[m,n] = x[m,:] . w_qkv[n,:]   (M=8192, N=2304, K=768)
// column n = h*192 + d*3 + s  (s: 0=q,1=k,2=v);  Q gets SCALE*LOG2E folded in.

__global__ void __launch_bounds__(128) mha_qkv(const half_t* __restrict__ xh,
                                               const half_t* __restrict__ wqh,
                                               half_t* __restrict__ qh,
                                               half_t* __restrict__ kh,
                                               half_t* __restrict__ vh) {
    const int lane = threadIdx.x & 31;
    const int wave = blockIdx.x * (blockDim.x >> 5) + (threadIdx.x >> 5);
    const int nTn = (3 * DIM) / 64;              // 36
    const int tm = (wave / nTn) * 16;
    const int tn = (wave % nTn) * 64;

    v8f acc[4] = {};
    const int arow = tm + (lane & 15);
#pragma unroll 4
    for (int k0 = 0; k0 < DIM; k0 += 32) {
        v16h a = loadA(xh, DIM, arow, k0, lane);
#pragma unroll
        for (int t = 0; t < 4; ++t) {
            v16h b = loadBT(wqh, DIM, tn + t * 16 + (lane & 15), k0, lane);
            acc[t] = WMMA(a, b, acc[t]);
        }
    }
    const int hid = lane >> 4;
#pragma unroll
    for (int r = 0; r < 8; ++r) {
        int m  = tm + r + 8 * hid;
        int bi = m >> 10, ni = m & (Nn - 1);
#pragma unroll
        for (int t = 0; t < 4; ++t) {
            int col = tn + t * 16 + (lane & 15);
            int s = col % 3, hd = col / 3;       // hd = h*64 + d
            int d = hd & 63, hh = hd >> 6;
            size_t o = (((size_t)bi * H + hh) * Nn + ni) * Dh + d;
            float val = acc[t][r];
            if (s == 0)      qh[o] = (half_t)(val * (SCALE * LOG2E));
            else if (s == 1) kh[o] = (half_t)val;
            else             vh[o] = (half_t)val;
        }
    }
}

// ---- kernel 2: flash attention per (b,h,64-query block) ---------------------
// 4 waves; wave owns 16 query rows. K/V 64x64 tiles double-buffered in LDS via
// async global->LDS DMA (ASYNCcnt); online softmax in exp2 domain; P staged
// through LDS for the C->A layout reshuffle.

// one async b128 chunk: LDS[ldsOff] = MEM[gaddr .. gaddr+15], tracked by ASYNCcnt
__device__ __forceinline__ void async_b128(unsigned int ldsOff, const void* gaddr) {
    asm volatile("global_load_async_to_lds_b128 %0, %1, off"
                 :: "v"(ldsOff), "v"(gaddr) : "memory");
}

// copy one 64x64 f16 K tile + V tile (8KB each) into LDS: 8 async ops / thread
__device__ __forceinline__ void async_tile_load(unsigned int ldsK, unsigned int ldsV,
                                                const half_t* K, const half_t* V,
                                                int kc, int tid) {
#pragma unroll
    for (int j = 0; j < 4; ++j) {
        int i = tid + j * 128;                           // 512 x 16B chunks
        async_b128(ldsK + i * 16, (const void*)(K + (size_t)kc * Dh + i * 8));
        async_b128(ldsV + i * 16, (const void*)(V + (size_t)kc * Dh + i * 8));
    }
}

__global__ void __launch_bounds__(128) mha_attn(const half_t* __restrict__ qh,
                                                const half_t* __restrict__ kh,
                                                const half_t* __restrict__ vh,
                                                half_t* __restrict__ oh) {
    __shared__ half_t sK[2][64 * 64];
    __shared__ half_t sV[2][64 * 64];
    __shared__ half_t sP[4][16 * 64];

    const int lane = threadIdx.x & 31;
    const int wv   = threadIdx.x >> 5;
    const int bh   = blockIdx.x >> 4;            // 96 (b,h) pairs
    const int qblk = (blockIdx.x & 15) * 64;
    const int bi = bh / H, hi = bh % H;

    const half_t* Q = qh + (size_t)bh * Nn * Dh;
    const half_t* K = kh + (size_t)bh * Nn * Dh;
    const half_t* V = vh + (size_t)bh * Nn * Dh;

    // workgroup-relative LDS byte offsets (generic LDS ptr low 32 bits)
    const unsigned int offK0 = (unsigned int)(uintptr_t)&sK[0][0];
    const unsigned int offK1 = (unsigned int)(uintptr_t)&sK[1][0];
    const unsigned int offV0 = (unsigned int)(uintptr_t)&sV[0][0];
    const unsigned int offV1 = (unsigned int)(uintptr_t)&sV[1][0];

    const int qrow = qblk + wv * 16;
    const v16h aQ0 = loadA(Q, Dh, qrow + (lane & 15), 0,  lane);
    const v16h aQ1 = loadA(Q, Dh, qrow + (lane & 15), 32, lane);

    float mrun[8], lrun[8];
    v8f accO[4] = {};
#pragma unroll
    for (int r = 0; r < 8; ++r) { mrun[r] = -1e30f; lrun[r] = 0.f; }

    // prologue: DMA tile 0 into buffer 0
    async_tile_load(offK0, offV0, K, V, 0, threadIdx.x);

    for (int it = 0; it < Nn / 64; ++it) {
        const int buf = it & 1;
        if (it + 1 < Nn / 64) {
            // prefetch next tile into the other buffer, then wait only for the
            // PREVIOUS tile's 8 ops (async ops complete in order)
            async_tile_load(buf ? offK0 : offK1, buf ? offV0 : offV1,
                            K, V, (it + 1) * 64, threadIdx.x);
            asm volatile("s_wait_asynccnt 0x8" ::: "memory");
        } else {
            asm volatile("s_wait_asynccnt 0x0" ::: "memory");
        }
        __syncthreads();
        const half_t* cK = &sK[buf][0];
        const half_t* cV = &sV[buf][0];

        // S = Q . K^T  (B[d][k] = cK[k][d]: d contiguous -> loadBT)
        v8f S[4];
#pragma unroll
        for (int t = 0; t < 4; ++t) {
            v16h b0 = loadBT(cK, Dh, t * 16 + (lane & 15), 0,  lane);
            v16h b1 = loadBT(cK, Dh, t * 16 + (lane & 15), 32, lane);
            v8f c = {};
            c = WMMA(aQ0, b0, c);
            c = WMMA(aQ1, b1, c);
            S[t] = c;
        }

        // online softmax: rows live in (vgpr r, half); reduce across 16 lanes
        float pv[4][8];
#pragma unroll
        for (int r = 0; r < 8; ++r) {
            float mx = fmaxf(fmaxf(S[0][r], S[1][r]), fmaxf(S[2][r], S[3][r]));
#pragma unroll
            for (int off = 1; off < 16; off <<= 1)
                mx = fmaxf(mx, __shfl_xor(mx, off, 32));
            float mnew  = fmaxf(mrun[r], mx);
            float alpha = __builtin_amdgcn_exp2f(mrun[r] - mnew);
            float ss = 0.f;
#pragma unroll
            for (int t = 0; t < 4; ++t) {
                float p = __builtin_amdgcn_exp2f(S[t][r] - mnew);
                pv[t][r] = p; ss += p;
            }
#pragma unroll
            for (int off = 1; off < 16; off <<= 1)
                ss += __shfl_xor(ss, off, 32);
            lrun[r] = lrun[r] * alpha + ss;
            mrun[r] = mnew;
#pragma unroll
            for (int t = 0; t < 4; ++t) accO[t][r] *= alpha;
        }

        // stage P (C-layout -> row-major LDS -> A-layout)
        half_t* P = &sP[wv][0];
        const int hid = lane >> 4;
#pragma unroll
        for (int r = 0; r < 8; ++r) {
            int rr = r + 8 * hid;
#pragma unroll
            for (int t = 0; t < 4; ++t)
                P[rr * 64 + t * 16 + (lane & 15)] = (half_t)pv[t][r];
        }
        asm volatile("s_wait_dscnt 0x0" ::: "memory");  // wave-local LDS RAW

        // O += P . V  (contraction over 64 keys; V row-major in k -> strided B)
        v16h aP0 = loadA(P, 64, (lane & 15), 0,  lane);
        v16h aP1 = loadA(P, 64, (lane & 15), 32, lane);
#pragma unroll
        for (int t = 0; t < 4; ++t) {
            v16h b0 = loadBS(cV, Dh, 0,  t * 16 + (lane & 15), lane);
            v16h b1 = loadBS(cV, Dh, 32, t * 16 + (lane & 15), lane);
            accO[t] = WMMA(aP0, b0, accO[t]);
            accO[t] = WMMA(aP1, b1, accO[t]);
        }
        __syncthreads();   // all reads of this buffer done before it+2 rewrites
    }

    // normalize, store to attention buffer [B*N, DIM] f16 (col = h*64 + d)
    const int hid = lane >> 4;
#pragma unroll
    for (int r = 0; r < 8; ++r) {
        float inv = 1.f / lrun[r];
        int m = bi * Nn + qrow + r + 8 * hid;
#pragma unroll
        for (int t = 0; t < 4; ++t) {
            int col = hi * Dh + t * 16 + (lane & 15);
            oh[(size_t)m * DIM + col] = (half_t)(accO[t][r] * inv);
        }
    }
}

// ---- kernel 3: output projection + bias, fp32 out ---------------------------

__global__ void __launch_bounds__(128) mha_proj(const half_t* __restrict__ ah,
                                                const half_t* __restrict__ wph,
                                                const float* __restrict__ bias,
                                                float* __restrict__ out) {
    const int lane = threadIdx.x & 31;
    const int wave = blockIdx.x * (blockDim.x >> 5) + (threadIdx.x >> 5);
    const int nTn = DIM / 64;                    // 12
    const int tm = (wave / nTn) * 16;
    const int tn = (wave % nTn) * 64;

    v8f acc[4] = {};
    const int arow = tm + (lane & 15);
#pragma unroll 4
    for (int k0 = 0; k0 < DIM; k0 += 32) {
        v16h a = loadA(ah, DIM, arow, k0, lane);
#pragma unroll
        for (int t = 0; t < 4; ++t) {
            v16h b = loadBT(wph, DIM, tn + t * 16 + (lane & 15), k0, lane);
            acc[t] = WMMA(a, b, acc[t]);
        }
    }
    const int hid = lane >> 4;
#pragma unroll
    for (int r = 0; r < 8; ++r) {
        int m = tm + r + 8 * hid;
#pragma unroll
        for (int t = 0; t < 4; ++t) {
            int col = tn + t * 16 + (lane & 15);
            out[(size_t)m * DIM + col] = acc[t][r] + bias[col];
        }
    }
}

// ---- launch -----------------------------------------------------------------

extern "C" void kernel_launch(void* const* d_in, const int* in_sizes, int n_in,
                              void* d_out, int out_size, void* d_ws, size_t ws_size,
                              hipStream_t stream) {
    const float* x      = (const float*)d_in[0];
    const float* w_qkv  = (const float*)d_in[1];
    const float* w_proj = (const float*)d_in[2];
    const float* b_proj = (const float*)d_in[3];
    float* out = (float*)d_out;

    char* ws = (char*)d_ws;
    size_t off = 0;
    auto carve = [&](size_t bytes) {
        void* p = ws + off;
        off += (bytes + 255) & ~(size_t)255;
        return p;
    };
    half_t* xh  = (half_t*)carve((size_t)Bb * Nn * DIM * 2);
    half_t* wqh = (half_t*)carve((size_t)3 * DIM * DIM * 2);
    half_t* wph = (half_t*)carve((size_t)DIM * DIM * 2);
    half_t* qh  = (half_t*)carve((size_t)Bb * H * Nn * Dh * 2);
    half_t* kh  = (half_t*)carve((size_t)Bb * H * Nn * Dh * 2);
    half_t* vh  = (half_t*)carve((size_t)Bb * H * Nn * Dh * 2);
    half_t* ah  = (half_t*)carve((size_t)Bb * Nn * DIM * 2);

    mha_cvt<<<4096, 256, 0, stream>>>(x, w_qkv, w_proj, xh, wqh, wph);

    // QKV: (8192/16)*(2304/64) = 18432 waves, 4 waves/block
    mha_qkv<<<4608, 128, 0, stream>>>(xh, wqh, qh, kh, vh);

    // attention: 96 (b,h) * 16 query blocks
    mha_attn<<<1536, 128, 0, stream>>>(qh, kh, vh, ah);

    // projection: (8192/16)*(768/64) = 6144 waves, 4 waves/block
    mha_proj<<<1536, 128, 0, stream>>>(ah, wph, b_proj, out);
}
